// Ptr_Layer_9414568312956
// MI455X (gfx1250) — matrix operand order repaired and checked by reading, not compiled, not run on hardware
//
#include <hip/hip_runtime.h>

// Problem constants (match reference)
#define NBATCH 16
#define BN     2048
#define DD     128

#define BK     32            // keys per iteration
#define WAVES  8
#define BQ     (16 * WAVES)  // 128 queries per workgroup (16 per wave)

typedef __attribute__((ext_vector_type(16))) __bf16 v16bf;
typedef __attribute__((ext_vector_type(8)))  __bf16 v8bf;
typedef __attribute__((ext_vector_type(8)))  float  v8f;

#define SE_STRIDE  136   // bf16 per row: 128 + 8 pad (272B, 16B aligned)
#define SXT_STRIDE 40    // bf16 per row: 32 + 8 pad  (80B, 16B aligned)

#define LOG2E 1.4426950408889634f

__device__ __forceinline__ v16bf ld_frag(const __bf16* p0, const __bf16* p1) {
  v8bf lo = *(const v8bf*)p0;
  v8bf hi = *(const v8bf*)p1;
  return __builtin_shufflevector(lo, hi, 0,1,2,3,4,5,6,7,8,9,10,11,12,13,14,15);
}

// Stage one 32-key tile: e/W2 row-major bf16, x transposed bf16.
// 256 threads: 32 rows x (8 threads x 16 contiguous floats).
__device__ __forceinline__ void stage_tile(__bf16* __restrict__ dE,
                                           __bf16* __restrict__ dW2,
                                           __bf16* __restrict__ dXT,
                                           const float* __restrict__ eb,
                                           const float* __restrict__ xb,
                                           const float* __restrict__ W2,
                                           int mbase, int srow, int scol,
                                           bool prefetch_next)
{
  const float* ep = eb + (size_t)(mbase + srow) * DD + scol;
  const float* wp = W2 + (size_t)(mbase + srow) * DD + scol;
  const float* xp = xb + (size_t)(mbase + srow) * DD + scol;
  #pragma unroll
  for (int j = 0; j < 16; ++j) {
    dE [srow * SE_STRIDE + scol + j]    = (__bf16)ep[j];
    dW2[srow * SE_STRIDE + scol + j]    = (__bf16)wp[j];
    dXT[(scol + j) * SXT_STRIDE + srow] = (__bf16)xp[j];
  }
  if (prefetch_next) {   // hint tile it+2 into cache (global_prefetch_b8)
    __builtin_prefetch(ep + (size_t)BK * DD, 0, 1);
    __builtin_prefetch(wp + (size_t)BK * DD, 0, 1);
    __builtin_prefetch(xp + (size_t)BK * DD, 0, 1);
  }
}

__global__ __launch_bounds__(256)
void fused_tanh_softsort_attn(const float* __restrict__ x,
                              const float* __restrict__ e,
                              const float* __restrict__ d,
                              const float* __restrict__ W1,
                              const float* __restrict__ W2,
                              float* __restrict__ out)
{
  // Double-buffered tiles: 2 x (8.5 + 8.5 + 10) KB = 55.3 KB LDS
  __shared__ __align__(16) __bf16 sE [2][BK * SE_STRIDE];   // e tile  [key][d]
  __shared__ __align__(16) __bf16 sW2[2][BK * SE_STRIDE];   // W2 tile [key][d]
  __shared__ __align__(16) __bf16 sXT[2][DD * SXT_STRIDE];  // x tile^T [feat][key]

  const int b     = blockIdx.x / (BN / BQ);
  const int qtile = blockIdx.x % (BN / BQ);
  const int wv    = threadIdx.x >> 5;
  const int lane  = threadIdx.x & 31;
  const int n     = lane & 15;   // column: query (scores) / feature (output)
  const int ch    = lane >> 4;   // lane half
  const int qbase = qtile * BQ + wv * 16;

  // ---- loop-invariant B-side fragments: W1 rows / d rows of this wave's 16 queries
  // B[K=dcol][N=query]: lane element j = row[32*dc + 16*ch + j]
  v16bf bW1[4], bD[4];
  {
    const float* w1r = W1 + (size_t)(qbase + n) * DD;
    const float* dr  = d  + ((size_t)b * BN + qbase + n) * DD;
    #pragma unroll
    for (int dc = 0; dc < 4; ++dc) {
      const int col = 32 * dc + 16 * ch;
      #pragma unroll
      for (int j = 0; j < 16; ++j) {
        bW1[dc][j] = (__bf16)w1r[col + j];
        bD [dc][j] = (__bf16)dr [col + j];
      }
    }
  }

  v8f acc[8];
  #pragma unroll
  for (int ft = 0; ft < 8; ++ft) acc[ft] = (v8f){};
  float l_run = 0.0f;

  const float* eb = e + (size_t)b * BN * DD;
  const float* xb = x + (size_t)b * BN * DD;

  const int srow = threadIdx.x >> 3;         // staging: 32 rows, 8 threads/row
  const int scol = (threadIdx.x & 7) * 16;

  // ---- prologue: stage tile 0
  stage_tile(sE[0], sW2[0], sXT[0], eb, xb, W2, 0, srow, scol, true);
  __syncthreads();

  const int NIT = BN / BK;  // 64
  for (int it = 0; it < NIT; ++it) {
    const int cur = it & 1;
    const __bf16* cE  = sE [cur];
    const __bf16* cW2 = sW2[cur];
    const __bf16* cXT = sXT[cur];

    // ---- scores: D[M=key, N=query] = e@W1^T + W2@d^T, two 16x16 tiles over 32 keys
    v8f s0 = (v8f){}, s1 = (v8f){};
    #pragma unroll
    for (int dc = 0; dc < 4; ++dc) {
      const int c0 = 32 * dc + 16 * ch;  // A-fragment K contiguous: k = 16*ch + j
      const int c1 = c0 + 8;
      const __bf16* e0 = &cE [( 0 + n) * SE_STRIDE];
      const __bf16* e1 = &cE [(16 + n) * SE_STRIDE];
      const __bf16* w0 = &cW2[( 0 + n) * SE_STRIDE];
      const __bf16* w1 = &cW2[(16 + n) * SE_STRIDE];
      v16bf ae0 = ld_frag(e0 + c0, e0 + c1);
      v16bf ae1 = ld_frag(e1 + c0, e1 + c1);
      v16bf aw0 = ld_frag(w0 + c0, w0 + c1);
      v16bf aw1 = ld_frag(w1 + c0, w1 + c1);
      s0 = __builtin_amdgcn_wmma_f32_16x16x32_bf16(false, ae0, false, bW1[dc], (short)0, s0, false, false);
      s0 = __builtin_amdgcn_wmma_f32_16x16x32_bf16(false, aw0, false, bD[dc],  (short)0, s0, false, false);
      s1 = __builtin_amdgcn_wmma_f32_16x16x32_bf16(false, ae1, false, bW1[dc], (short)0, s1, false, false);
      s1 = __builtin_amdgcn_wmma_f32_16x16x32_bf16(false, aw1, false, bD[dc],  (short)0, s1, false, false);
    }

    // ---- tanh + exp(s-1): tanh<=1 so a constant shift gives an exact, stable softmax.
    // tanh(x) = 1 - 2*rcp(e^{2x}+1)  (NaN-free at both infinities)
    // Score D-layout read back as PV A-layout is an identity per lane -> pack apv directly.
    v16bf apv;
    float ls = 0.0f;
    #pragma unroll
    for (int r = 0; r < 8; ++r) {
      float t0  = __builtin_amdgcn_exp2f(s0[r] * (2.0f * LOG2E));
      float t1  = __builtin_amdgcn_exp2f(s1[r] * (2.0f * LOG2E));
      float th0 = 1.0f - 2.0f * __builtin_amdgcn_rcpf(t0 + 1.0f);
      float th1 = 1.0f - 2.0f * __builtin_amdgcn_rcpf(t1 + 1.0f);
      float p0  = __builtin_amdgcn_exp2f((th0 - 1.0f) * LOG2E);
      float p1  = __builtin_amdgcn_exp2f((th1 - 1.0f) * LOG2E);
      ls += p0 + p1;
      apv[r]     = (__bf16)p0;   // keys  8*ch + r       (tile 0)
      apv[r + 8] = (__bf16)p1;   // keys 16 + 8*ch + r   (tile 1)
    }
    l_run += ls + __shfl_xor(ls, 16, 32);  // combine lane-half partials per query

    // ---- PV: acc[ft] (D[M=query, N=feature]) += P^T (16q x 32k) @ X (32k x 16f)
    #pragma unroll
    for (int ft = 0; ft < 8; ++ft) {
      const __bf16* xr = &cXT[(16 * ft + n) * SXT_STRIDE + 8 * ch];
      v16bf bx = ld_frag(xr, xr + 16);
      acc[ft] = __builtin_amdgcn_wmma_f32_16x16x32_bf16(false, apv, false, bx, (short)0, acc[ft], false, false);
    }

    // ---- stage tile it+1 into the other buffer while WMMAs drain.
    // That buffer was last read in iter it-1, before the previous barrier.
    if (it + 1 < NIT) {
      stage_tile(sE[cur ^ 1], sW2[cur ^ 1], sXT[cur ^ 1],
                 eb, xb, W2, (it + 1) * BK, srow, scol, it + 2 < NIT);
    }
    __syncthreads();  // single barrier per iteration
  }

  // ---- normalize by softmax denominator and store
  float inv = __builtin_amdgcn_rcpf(l_run);   // lane holds denom of query n
  float invq[8];
  #pragma unroll
  for (int r = 0; r < 8; ++r) invq[r] = __shfl(inv, r + 8 * ch, 32);  // to out layout

  #pragma unroll
  for (int ft = 0; ft < 8; ++ft) {
    #pragma unroll
    for (int r = 0; r < 8; ++r) {
      out[((size_t)b * BN + qbase + r + 8 * ch) * DD + 16 * ft + n] = acc[ft][r] * invq[r];
    }
  }
}

extern "C" void kernel_launch(void* const* d_in, const int* in_sizes, int n_in,
                              void* d_out, int out_size, void* d_ws, size_t ws_size,
                              hipStream_t stream) {
  (void)in_sizes; (void)n_in; (void)out_size; (void)d_ws; (void)ws_size;
  const float* x  = (const float*)d_in[0];
  const float* e  = (const float*)d_in[1];
  const float* dd = (const float*)d_in[2];
  const float* W1 = (const float*)d_in[3];
  const float* W2 = (const float*)d_in[4];
  float* out = (float*)d_out;

  dim3 grid(NBATCH * (BN / BQ));  // 16 batches x 16 query tiles = 256 WGs
  dim3 block(256);                // 8 waves of 32
  fused_tanh_softsort_attn<<<grid, block, 0, stream>>>(x, e, dd, W1, W2, out);
}
